// GraphGen_71631464562749
// MI455X (gfx1250) — matrix operand order repaired and checked by reading, not compiled, not run on hardware
//
#include <hip/hip_runtime.h>
#include <hip/hip_bf16.h>
#include <math.h>

// Problem constants (from reference)
#define BS    64
#define NPTS  512
#define FEAT  64
#define HID   128
#define ITERS 3
#define MTOT  (BS * NPTS)          // 32768 total rows
#define TEMP_INV 100.0f            // 1 / 0.01

typedef __attribute__((ext_vector_type(16))) __bf16 v16bf;
typedef __attribute__((ext_vector_type(8)))  float  v8f;

// ---------------------------------------------------------------------------
// WMMA helpers (wave32; one wave owns 16x16 f32 tiles, K stepped by 32)
// ---------------------------------------------------------------------------

static __device__ __forceinline__ v8f wmma_bf16(v16bf a, v16bf b, v8f c) {
  return __builtin_amdgcn_wmma_f32_16x16x32_bf16(
      /*neg_a=*/false, a, /*neg_b=*/false, b,
      /*c_mod=*/(short)0, c, /*reuse_a=*/false, /*reuse_b=*/false);
}

// A-matrix 16x32 bf16 fragment from row-major f32 [.. x ld], tile @ (row0,k0).
// Lanes 0-15: K 0..15 ; lanes 16-31: K 16..31 (matches swizzled B order).
// Contiguous 8-float halves -> vectorizes to global_load_b128 + v_cvt_pk_bf16.
static __device__ __forceinline__ v16bf
load_a_frag(const float* __restrict__ src, int ld, int row0, int k0, int lane) {
  const float* p = src + (size_t)(row0 + (lane & 15)) * ld + k0 + ((lane >> 4) << 4);
  v16bf a;
#pragma unroll
  for (int j = 0; j < 8; ++j) {
    a[j]     = (__bf16)p[j];
    a[8 + j] = (__bf16)p[8 + j];
  }
  return a;
}

// Swizzled B layout: element (k,n) of a [K x 128] B operand lives at
//   ((s*8 + c)*32 + lane)*16 + j,  s=k>>5, c=n>>4, lane=(n&15)+16*((k&31)>>4), j=k&15
// so one lane's 16 bf16 fragment values are 32 contiguous bytes.
static __device__ __forceinline__ int swz_idx(int k, int n) {
  int s = k >> 5, kk = k & 31, c = n >> 4;
  int lane = (n & 15) + ((kk >> 4) << 4);
  return (((s * 8 + c) * 32 + lane) << 4) + (kk & 15);
}

// B-fragment load from swizzled bf16 buffer: 32B contiguous -> 2x b128 loads.
static __device__ __forceinline__ v16bf
load_b_swz(const __bf16* __restrict__ wsz, int k0, int col0, int lane) {
  return *(const v16bf*)(wsz + (((((k0 >> 5) * 8) + (col0 >> 4)) * 32 + lane) << 4));
}

static __device__ __forceinline__ float sigmoidf(float x) {
  return 1.0f / (1.0f + __expf(-x));
}

// ---------------------------------------------------------------------------
// One-time weight swizzle: f32 row-major [K x 128] -> bf16 fragment layout
// ---------------------------------------------------------------------------
__global__ __launch_bounds__(256) void swizzle_kernel(
    const float* __restrict__ src, __bf16* __restrict__ dst, int K) {
  int id = blockIdx.x * blockDim.x + threadIdx.x;
  if (id >= K * HID) return;
  int k = id / HID, n = id % HID;
  dst[swz_idx(k, n)] = (__bf16)src[id];
}

// ---------------------------------------------------------------------------
// h = relu(x @ W_emb + b_emb) + pos_table[row % N]      [32768 x 128], K=64
// ---------------------------------------------------------------------------
__global__ __launch_bounds__(128) void embed_kernel(
    const float* __restrict__ x, const __bf16* __restrict__ Wsz,
    const float* __restrict__ bias, const float* __restrict__ pos,
    float* __restrict__ h) {
  int lane = threadIdx.x & 31;
  int tile = blockIdx.x * 4 + (threadIdx.x >> 5);
  int row0 = (tile >> 3) << 4;
  int col0 = (tile & 7) << 4;
  v8f acc = {};
#pragma unroll
  for (int k0 = 0; k0 < FEAT; k0 += 32) {
    acc = wmma_bf16(load_a_frag(x, FEAT, row0, k0, lane),
                    load_b_swz(Wsz, k0, col0, lane), acc);
  }
  int n     = col0 + (lane & 15);
  int rbase = row0 + ((lane >> 4) << 3);
  float bn  = bias[n];
#pragma unroll
  for (int v = 0; v < 8; ++v) {
    int r = rbase + v;
    h[(size_t)r * HID + n] =
        fmaxf(acc[v] + bn, 0.0f) + pos[(size_t)(r & (NPTS - 1)) * HID + n];
  }
}

// ---------------------------------------------------------------------------
// m = relu(h @ Wm + bm), emitted directly in swizzled bf16 B-fragment layout
// (m is only ever the B operand of agg) -> halves store traffic, kills gathers
// ---------------------------------------------------------------------------
__global__ __launch_bounds__(128) void msg_kernel(
    const float* __restrict__ h, const __bf16* __restrict__ Wmsz,
    const float* __restrict__ bm, __bf16* __restrict__ mswz) {
  int lane = threadIdx.x & 31;
  int tile = blockIdx.x * 4 + (threadIdx.x >> 5);
  int row0 = (tile >> 3) << 4;
  int col0 = (tile & 7) << 4;
  v8f acc = {};
#pragma unroll
  for (int k0 = 0; k0 < HID; k0 += 32) {
    acc = wmma_bf16(load_a_frag(h, HID, row0, k0, lane),
                    load_b_swz(Wmsz, k0, col0, lane), acc);
  }
  int n     = col0 + (lane & 15);
  int rbase = row0 + ((lane >> 4) << 3);
  float bn  = bm[n];
#pragma unroll
  for (int v = 0; v < 8; ++v) {
    int r  = rbase + v;
    int bb = r >> 9;             // batch
    int k  = r & (NPTS - 1);     // row within batch == K index of agg's B operand
    mswz[(size_t)bb * (NPTS * HID) + swz_idx(k, n)] =
        (__bf16)fmaxf(acc[v] + bn, 0.0f);
  }
}

// ---------------------------------------------------------------------------
// agg[b] = A[b] @ m[b]   (512x512 @ 512x128, K=512 -- the big GEMM)
// 256-thread blocks (8 waves). Per K-step the block async-copies the 8KB
// swizzled m slab into LDS (global_load_async_to_lds_b128, ASYNCcnt), then
// each wave runs 8 WMMAs off one A-fragment (16 rows x full 128 cols).
// ---------------------------------------------------------------------------
__global__ __launch_bounds__(256) void agg_kernel(
    const float* __restrict__ A, const __bf16* __restrict__ mswz,
    float* __restrict__ agg) {
  __shared__ __bf16 lds_m[32 * HID];          // 4096 bf16 = 8 KB per K-step slab
  int tid  = threadIdx.x;
  int lane = tid & 31;
  int wave = tid >> 5;                        // 0..7
  int b    = blockIdx.x >> 2;                 // 64 batches
  int row0 = ((blockIdx.x & 3) << 7) + (wave << 4);
  const float*  Ab = A    + (size_t)b * NPTS * NPTS;
  const __bf16* mb = mswz + (size_t)b * NPTS * HID;

  v8f acc[8] = {};
  unsigned lds_dst = (unsigned)(uintptr_t)lds_m + (unsigned)tid * 32u;

  for (int k0 = 0; k0 < NPTS; k0 += 32) {
    // Stage this K-step's 8KB B slab: 256 lanes x 32B, async to LDS.
    {
      const __bf16* gp = mb + ((size_t)(k0 >> 5) * (32 * HID)) + tid * 16;
      unsigned long long gaddr = (unsigned long long)(uintptr_t)gp;
      asm volatile(
          "global_load_async_to_lds_b128 %0, %1, off\n\t"
          "global_load_async_to_lds_b128 %0, %1, off offset:16"
          :: "v"(lds_dst), "v"(gaddr) : "memory");
      asm volatile("s_wait_asynccnt 0x0" ::: "memory");
    }
    __syncthreads();

    v16bf fa = load_a_frag(Ab, NPTS, row0, k0, lane);   // shared by 8 WMMAs
#pragma unroll
    for (int c = 0; c < 8; ++c) {
      v16bf fb = *(const v16bf*)(lds_m + (((c * 32) + lane) << 4));
      acc[c] = wmma_bf16(fa, fb, acc[c]);
    }
    __syncthreads();
  }

  float* out = agg + (size_t)b * NPTS * HID;
  int rbase = row0 + ((lane >> 4) << 3);
  int nn    = lane & 15;
#pragma unroll
  for (int c = 0; c < 8; ++c)
#pragma unroll
    for (int v = 0; v < 8; ++v)
      out[(size_t)(rbase + v) * HID + (c << 4) + nn] = acc[c][v];
}

// ---------------------------------------------------------------------------
// GRU phase 1 (fused gates): Z = sig(agg@Wz + h@Uz + bz), R = sig(...),
// P = agg@Wh + bh, RH = R*h.  5 WMMAs per K-step sharing two A-fragments;
// all weights come pre-swizzled -> B loads are 2x b128 each.
// ---------------------------------------------------------------------------
__global__ __launch_bounds__(128) void gru1_kernel(
    const float* __restrict__ agg, const float* __restrict__ h,
    const __bf16* __restrict__ Wz, const __bf16* __restrict__ Uz, const float* __restrict__ bz,
    const __bf16* __restrict__ Wr, const __bf16* __restrict__ Ur, const float* __restrict__ br,
    const __bf16* __restrict__ Wh, const float* __restrict__ bh,
    float* __restrict__ Z, float* __restrict__ P, float* __restrict__ RH) {
  int lane = threadIdx.x & 31;
  int tile = blockIdx.x * 4 + (threadIdx.x >> 5);
  int row0 = (tile >> 3) << 4;
  int col0 = (tile & 7) << 4;
  v8f az = {}, ar = {}, ap = {};
#pragma unroll
  for (int k0 = 0; k0 < HID; k0 += 32) {
    v16bf fa = load_a_frag(agg, HID, row0, k0, lane);
    v16bf fh = load_a_frag(h,   HID, row0, k0, lane);
    az = wmma_bf16(fa, load_b_swz(Wz, k0, col0, lane), az);
    az = wmma_bf16(fh, load_b_swz(Uz, k0, col0, lane), az);
    ar = wmma_bf16(fa, load_b_swz(Wr, k0, col0, lane), ar);
    ar = wmma_bf16(fh, load_b_swz(Ur, k0, col0, lane), ar);
    ap = wmma_bf16(fa, load_b_swz(Wh, k0, col0, lane), ap);
  }
  int n     = col0 + (lane & 15);
  int rbase = row0 + ((lane >> 4) << 3);
  float bzn = bz[n], brn = br[n], bhn = bh[n];
#pragma unroll
  for (int v = 0; v < 8; ++v) {
    size_t idx = (size_t)(rbase + v) * HID + n;
    float zv = sigmoidf(az[v] + bzn);
    float rv = sigmoidf(ar[v] + brn);
    Z[idx]  = zv;
    P[idx]  = ap[v] + bhn;
    RH[idx] = rv * h[idx];
  }
}

// ---------------------------------------------------------------------------
// GRU phase 2: hh = tanh(P + RH @ Uh);  h = (1-Z)*h + Z*hh   (in-place on h)
// ---------------------------------------------------------------------------
__global__ __launch_bounds__(128) void gru2_kernel(
    const float* __restrict__ RH, const __bf16* __restrict__ Uh,
    const float* __restrict__ P, const float* __restrict__ Z,
    float* __restrict__ h) {
  int lane = threadIdx.x & 31;
  int tile = blockIdx.x * 4 + (threadIdx.x >> 5);
  int row0 = (tile >> 3) << 4;
  int col0 = (tile & 7) << 4;
  v8f acc = {};
#pragma unroll
  for (int k0 = 0; k0 < HID; k0 += 32) {
    acc = wmma_bf16(load_a_frag(RH, HID, row0, k0, lane),
                    load_b_swz(Uh, k0, col0, lane), acc);
  }
  int n     = col0 + (lane & 15);
  int rbase = row0 + ((lane >> 4) << 3);
#pragma unroll
  for (int v = 0; v < 8; ++v) {
    size_t idx = (size_t)(rbase + v) * HID + n;
    float hh = tanhf(P[idx] + acc[v]);
    float zv = Z[idx];
    h[idx] = (1.0f - zv) * h[idx] + zv * hh;
  }
}

// ---------------------------------------------------------------------------
// Positional GRU (N=3 output: WMMA would waste 13/16 of the tile -> VALU)
// ---------------------------------------------------------------------------
__global__ __launch_bounds__(256) void pos_gru_kernel(
    const float* __restrict__ h, float* __restrict__ s,
    const float* __restrict__ Wpz, const float* __restrict__ Upz, const float* __restrict__ bpz,
    const float* __restrict__ Wpr, const float* __restrict__ Upr, const float* __restrict__ bpr,
    const float* __restrict__ Wph, const float* __restrict__ Uph, const float* __restrict__ bph) {
  int id = blockIdx.x * blockDim.x + threadIdx.x;
  if (id >= MTOT) return;
  const float* hr = h + (size_t)id * HID;
  float az[3] = {0.f, 0.f, 0.f}, ar[3] = {0.f, 0.f, 0.f}, ah[3] = {0.f, 0.f, 0.f};
  for (int k = 0; k < HID; ++k) {
    float hv = hr[k];
#pragma unroll
    for (int c = 0; c < 3; ++c) {
      az[c] += hv * Wpz[k * 3 + c];
      ar[c] += hv * Wpr[k * 3 + c];
      ah[c] += hv * Wph[k * 3 + c];
    }
  }
  float sv[3] = {s[id * 3 + 0], s[id * 3 + 1], s[id * 3 + 2]};
  float zs[3], rs[3];
#pragma unroll
  for (int c = 0; c < 3; ++c) {
    float uz = sv[0] * Upz[c] + sv[1] * Upz[3 + c] + sv[2] * Upz[6 + c];
    float ur = sv[0] * Upr[c] + sv[1] * Upr[3 + c] + sv[2] * Upr[6 + c];
    zs[c] = sigmoidf(az[c] + uz + bpz[c]);
    rs[c] = sigmoidf(ar[c] + ur + bpr[c]);
  }
  float rh[3] = {rs[0] * sv[0], rs[1] * sv[1], rs[2] * sv[2]};
#pragma unroll
  for (int c = 0; c < 3; ++c) {
    float uh = rh[0] * Uph[c] + rh[1] * Uph[3 + c] + rh[2] * Uph[6 + c];
    float ss = tanhf(ah[c] + uh + bph[c]);
    s[id * 3 + c] = (1.0f - zs[c]) * sv[c] + zs[c] * ss;
  }
}

// ---------------------------------------------------------------------------
// A = exp(-||s_i - s_j||^2 / temp) * mask       (elementwise, 16.7M elements)
// ---------------------------------------------------------------------------
__global__ __launch_bounds__(256) void adj_kernel(
    const float* __restrict__ s, const float* __restrict__ mask,
    float* __restrict__ A) {
  int id = blockIdx.x * blockDim.x + threadIdx.x;
  int bi = id >> 9;
  int j  = id & (NPTS - 1);
  int bj = (bi & ~(NPTS - 1)) | j;
  const float* si = s + (size_t)bi * 3;
  const float* sj = s + (size_t)bj * 3;
  float d0 = si[0] - sj[0];
  float d1 = si[1] - sj[1];
  float d2 = si[2] - sj[2];
  float dd = d0 * d0 + d1 * d1 + d2 * d2;
  A[id] = __expf(-dd * TEMP_INV) * mask[id];
}

__global__ __launch_bounds__(256) void init_s_kernel(float* __restrict__ s) {
  int id = blockIdx.x * blockDim.x + threadIdx.x;
  if (id >= MTOT) return;
  int n = id & (NPTS - 1);
  s[id * 3 + 0] = ((float)n - (float)(NPTS - 1) * 0.5f) / (float)NPTS;
  s[id * 3 + 1] = 0.0f;
  s[id * 3 + 2] = 0.0f;
}

// ---------------------------------------------------------------------------
// Launch
// ---------------------------------------------------------------------------
extern "C" void kernel_launch(void* const* d_in, const int* in_sizes, int n_in,
                              void* d_out, int out_size, void* d_ws, size_t ws_size,
                              hipStream_t stream) {
  (void)in_sizes; (void)n_in; (void)out_size; (void)ws_size;
  const float* x     = (const float*)d_in[0];
  const float* mask  = (const float*)d_in[1];
  const float* W_emb = (const float*)d_in[2];
  const float* b_emb = (const float*)d_in[3];
  const float* pos   = (const float*)d_in[4];
  const float* Wm  = (const float*)d_in[5];  const float* bm  = (const float*)d_in[6];
  const float* Wz  = (const float*)d_in[7];  const float* Uz  = (const float*)d_in[8];
  const float* bz  = (const float*)d_in[9];
  const float* Wr  = (const float*)d_in[10]; const float* Ur  = (const float*)d_in[11];
  const float* br  = (const float*)d_in[12];
  const float* Wh  = (const float*)d_in[13]; const float* Uh  = (const float*)d_in[14];
  const float* bh  = (const float*)d_in[15];
  const float* Wpz = (const float*)d_in[16]; const float* Upz = (const float*)d_in[17];
  const float* bpz = (const float*)d_in[18];
  const float* Wpr = (const float*)d_in[19]; const float* Upr = (const float*)d_in[20];
  const float* bpr = (const float*)d_in[21];
  const float* Wph = (const float*)d_in[22]; const float* Uph = (const float*)d_in[23];
  const float* bph = (const float*)d_in[24];
  float* A = (float*)d_out;   // [64, 512, 512] f32; stays hot in the 192MB L2

  // Workspace layout
  const size_t SZ = (size_t)MTOT * HID * sizeof(float);      // 16 MB
  char* w = (char*)d_ws;
  float*  h    = (float*)(w + 0 * SZ);
  float*  agg  = (float*)(w + 1 * SZ);
  float*  Z    = (float*)(w + 2 * SZ);
  float*  P    = (float*)(w + 3 * SZ);
  float*  RH   = (float*)(w + 4 * SZ);
  __bf16* mswz = (__bf16*)(w + 5 * SZ);                      // 8 MB (bf16 swizzled)
  float*  s    = (float*)(w + 5 * SZ + SZ / 2);              // 384 KB
  __bf16* wsz  = (__bf16*)(w + 5 * SZ + SZ / 2 + (1u << 20)); // swizzled weights
  const size_t WSLOT = (size_t)HID * HID;                    // 16384 elems / slot

  // One-time weight swizzle into bf16 fragment layout (22 small launches)
  __bf16* Wemb_s = wsz;
  swizzle_kernel<<<(FEAT * HID + 255) / 256, 256, 0, stream>>>(W_emb, Wemb_s, FEAT);
  const size_t HH = (size_t)HID * HID;
  for (int i = 0; i < ITERS; ++i) {
    __bf16* base = wsz + (1 + i * 7) * WSLOT;
    swizzle_kernel<<<64, 256, 0, stream>>>(Wm + i * HH, base + 0 * WSLOT, HID);
    swizzle_kernel<<<64, 256, 0, stream>>>(Wz + i * HH, base + 1 * WSLOT, HID);
    swizzle_kernel<<<64, 256, 0, stream>>>(Uz + i * HH, base + 2 * WSLOT, HID);
    swizzle_kernel<<<64, 256, 0, stream>>>(Wr + i * HH, base + 3 * WSLOT, HID);
    swizzle_kernel<<<64, 256, 0, stream>>>(Ur + i * HH, base + 4 * WSLOT, HID);
    swizzle_kernel<<<64, 256, 0, stream>>>(Wh + i * HH, base + 5 * WSLOT, HID);
    swizzle_kernel<<<64, 256, 0, stream>>>(Uh + i * HH, base + 6 * WSLOT, HID);
  }

  const dim3 wblk(128);
  const dim3 wgrd((MTOT / 16) * (HID / 16) / 4);   // 4096 blocks = 16384 tiles
  const int adj_blocks = (BS * NPTS * NPTS) / 256; // 65536

  embed_kernel<<<wgrd, wblk, 0, stream>>>(x, Wemb_s, b_emb, pos, h);
  init_s_kernel<<<MTOT / 256, 256, 0, stream>>>(s);
  adj_kernel<<<adj_blocks, 256, 0, stream>>>(s, mask, A);

  for (int i = 0; i < ITERS; ++i) {
    __bf16* base = wsz + (1 + i * 7) * WSLOT;
    msg_kernel<<<wgrd, wblk, 0, stream>>>(h, base + 0 * WSLOT, bm + i * HID, mswz);
    agg_kernel<<<256, 256, 0, stream>>>(A, mswz, agg);
    gru1_kernel<<<wgrd, wblk, 0, stream>>>(agg, h,
                                           base + 1 * WSLOT, base + 2 * WSLOT, bz + i * HID,
                                           base + 3 * WSLOT, base + 4 * WSLOT, br + i * HID,
                                           base + 5 * WSLOT, bh + i * HID,
                                           Z, P, RH);
    gru2_kernel<<<wgrd, wblk, 0, stream>>>(RH, base + 6 * WSLOT, P, Z, h);
    pos_gru_kernel<<<MTOT / 256, 256, 0, stream>>>(h, s, Wpz, Upz, bpz,
                                                   Wpr, Upr, bpr, Wph, Uph, bph);
    adj_kernel<<<adj_blocks, 256, 0, stream>>>(s, mask, A);
  }
}
